// TopoFilter_80049600463708
// MI455X (gfx1250) — compile-verified, matrix operand order checked by reading, not run on hardware
//
#include <hip/hip_runtime.h>
#include <hip/hip_bf16.h>
#include <math.h>

typedef _Float16 v16h __attribute__((ext_vector_type(16)));
typedef _Float16 v8h  __attribute__((ext_vector_type(8)));
typedef _Float16 v4h  __attribute__((ext_vector_type(4)));
typedef float    v8f  __attribute__((ext_vector_type(8)));

#define NPTS   8192     // B*S
#define DDIM   1024
#define MBLK   32       // rows per workgroup
#define NTILE  64       // cols per j-tile (4 N-tiles of 16)
#define KCHUNK 128      // B K-chunk staged in LDS
#define LDAF   1032     // A-resident row stride in halves (2064B, 16B aligned)
#define LDB    136      // B row stride in halves (272B, 16B aligned)
#define BIGF   1.0e30f
#define NCHUNKS ((NPTS / NTILE) * (DDIM / KCHUNK))   // 128 tiles * 8 chunks = 1024

#define A_BYTES  (MBLK * LDAF * 2)                   // 66048
#define B_BYTES  (NTILE * LDB * 2)                   // 17408
#define LDS_BYTES (A_BYTES + 2 * B_BYTES)            // 100864 (< 320KB/WG)

// ---- CDNA5 async global->LDS (guarded; fallback = batched register staging) ----
#if defined(__has_builtin)
#if __has_builtin(__builtin_amdgcn_global_load_async_to_lds_b128)
#define HAVE_ASYNC_LDS 1
#endif
#endif
#ifndef HAVE_ASYNC_LDS
#define HAVE_ASYNC_LDS 0
#endif

// builtin signature: (int4 addrspace(1)*, int4 addrspace(3)*, imm offset, imm cpol)
typedef int v4i_t __attribute__((vector_size(16)));
typedef __attribute__((address_space(1))) v4i_t* glb_b128_t;
typedef __attribute__((address_space(3))) v4i_t* lds_b128_t;

__device__ __forceinline__ void wait_async_zero() {
#if HAVE_ASYNC_LDS
#if __has_builtin(__builtin_amdgcn_s_wait_asynccnt)
  __builtin_amdgcn_s_wait_asynccnt(0);
#else
  asm volatile("s_wait_asynccnt 0x0" ::: "memory");
#endif
#endif
}

// ---------------------------------------------------------------------------
// Kernel 1: row squared norms (fp32) + fp32 -> fp16 conversion of x_real
// ---------------------------------------------------------------------------
__global__ __launch_bounds__(256) void topo_prep_kernel(
    const float* __restrict__ xr, _Float16* __restrict__ xh,
    float* __restrict__ sq) {
  __shared__ float red[256];
  const int row = blockIdx.x;
  const int tid = threadIdx.x;
  const float4 v = reinterpret_cast<const float4*>(xr)[row * 256 + tid];
  v4h hv;
  hv[0] = (_Float16)v.x; hv[1] = (_Float16)v.y;
  hv[2] = (_Float16)v.z; hv[3] = (_Float16)v.w;
  reinterpret_cast<v4h*>(xh)[row * 256 + tid] = hv;
  red[tid] = v.x * v.x + v.y * v.y + v.z * v.z + v.w * v.w;
  __syncthreads();
#pragma unroll
  for (int off = 128; off > 0; off >>= 1) {
    if (tid < off) red[tid] += red[tid + off];
    __syncthreads();
  }
  if (tid == 0) sq[row] = red[0];
}

// sorted-ascending 5-element insertion (t0 <= .. <= t4)
__device__ __forceinline__ void ins5(float v, float& t0, float& t1,
                                     float& t2, float& t3, float& t4) {
  if (v >= t4) return;
  if (v < t2) {
    if (v < t1) {
      if (v < t0) { t4 = t3; t3 = t2; t2 = t1; t1 = t0; t0 = v; }
      else        { t4 = t3; t3 = t2; t2 = t1; t1 = v; }
    } else        { t4 = t3; t3 = t2; t2 = v; }
  } else {
    if (v < t3)   { t4 = t3; t3 = v; }
    else          { t4 = v; }
  }
}

// ---------------------------------------------------------------------------
// Kernel 2: A resident in LDS, double-buffered async B staging, WMMA Gram,
// register-resident kNN, weights. 256 blocks x 256 threads (8 waves, 2Mx4N).
// ---------------------------------------------------------------------------
__global__ __launch_bounds__(256) void topo_knn_kernel(
    const _Float16* __restrict__ xh, const float* __restrict__ sq,
    float* __restrict__ wout) {
  __shared__ alignas(16) char lds_raw[LDS_BYTES];
  _Float16* const Asm = (_Float16*)(lds_raw);
  _Float16* const B0  = (_Float16*)(lds_raw + A_BYTES);
  _Float16* const B1  = (_Float16*)(lds_raw + A_BYTES + B_BYTES);

  const int tid   = threadIdx.x;
  const int lane  = tid & 31;
  const int wave  = tid >> 5;
  const int waveM = wave >> 2;               // 0..1
  const int waveN = wave & 3;                // 0..3
  const int h     = lane & 15;               // row/col within 16x16 tile
  const int hiSel = lane >> 4;               // 0/1 -> K group & C-row half
  const int hiK   = hiSel << 3;
  const int rowBase = blockIdx.x * MBLK;

  // per-thread loop-invariant staging offsets (B: 1024 v8h, 4 per thread)
  size_t bSrc[4];                            // halves offset into xh (j0 added per chunk)
  int    bDst[4];                            // halves offset into B buffer
#pragma unroll
  for (int i = 0; i < 4; ++i) {
    const int e = tid + i * 256, r = e >> 4, kk = (e & 15) << 3;
    bSrc[i] = (size_t)r * DDIM + kk;
    bDst[i] = r * LDB + kk;
  }

  // rows owned by this lane's accumulator slots r=0..7
  const int gRowBase = rowBase + waveM * 16 + hiSel * 8;
  float sqr[8];
#pragma unroll
  for (int r = 0; r < 8; ++r) sqr[r] = sq[gRowBase + r];

  // 8 register-resident top-5 lists (one per owned row, this lane's col slot)
  float l0[8], l1[8], l2[8], l3[8], l4[8];
#pragma unroll
  for (int r = 0; r < 8; ++r) {
    l0[r] = BIGF; l1[r] = BIGF; l2[r] = BIGF; l3[r] = BIGF; l4[r] = BIGF;
  }

  const int nl = waveN * 16 + h;             // column slot within j-tile
  float sc = sq[nl];                         // sq[col] for tile 0, pipelined below

  // ---- prologue: stage full A (32 rows x 1024 halves) + B chunk 0 ----
#if HAVE_ASYNC_LDS
#pragma unroll
  for (int i = 0; i < 16; ++i) {
    const int e = tid + i * 256;             // 0..4095
    const int r = e >> 7, kk = (e & 127) << 3;
    __builtin_amdgcn_global_load_async_to_lds_b128(
        (glb_b128_t)(xh + (size_t)(rowBase + r) * DDIM + kk),
        (lds_b128_t)(Asm + r * LDAF + kk), 0, 0);
  }
#pragma unroll
  for (int i = 0; i < 4; ++i)
    __builtin_amdgcn_global_load_async_to_lds_b128(
        (glb_b128_t)(xh + bSrc[i]), (lds_b128_t)(B0 + bDst[i]), 0, 0);
#else
  for (int i = 0; i < 16; ++i) {
    const int e = tid + i * 256;
    const int r = e >> 7, kk = (e & 127) << 3;
    *reinterpret_cast<v8h*>(&Asm[r * LDAF + kk]) =
        *reinterpret_cast<const v8h*>(xh + (size_t)(rowBase + r) * DDIM + kk);
  }
  {
    v8h tb[4];
#pragma unroll
    for (int i = 0; i < 4; ++i) tb[i] = *reinterpret_cast<const v8h*>(xh + bSrc[i]);
#pragma unroll
    for (int i = 0; i < 4; ++i) *reinterpret_cast<v8h*>(&B0[bDst[i]]) = tb[i];
  }
#endif
  wait_async_zero();
  __syncthreads();

  // loop-invariant fragment bases
  const _Float16* const pa = Asm + (waveM * 16 + h) * LDAF + hiK;
  const int bOff = (waveN * 16 + h) * LDB + hiK;

  v8f acc = {};
  for (int c = 0; c < NCHUNKS; ++c) {
    const int cur = c & 1;
    const _Float16* const pb = (cur ? B1 : B0) + bOff;

    if (c + 1 < NCHUNKS) {
      const int cn = c + 1;
      _Float16* const bNxt = cur ? B0 : B1;
      const size_t chunkOff =
          (size_t)(cn >> 3) * NTILE * DDIM + (size_t)(cn & 7) * KCHUNK;
#if HAVE_ASYNC_LDS
#pragma unroll
      for (int i = 0; i < 4; ++i)
        __builtin_amdgcn_global_load_async_to_lds_b128(
            (glb_b128_t)(xh + bSrc[i] + chunkOff),
            (lds_b128_t)(bNxt + bDst[i]), 0, 0);
#else
      v8h tb[4];
#pragma unroll
      for (int i = 0; i < 4; ++i)
        tb[i] = *reinterpret_cast<const v8h*>(xh + bSrc[i] + chunkOff);
#pragma unroll
      for (int i = 0; i < 4; ++i)
        *reinterpret_cast<v8h*>(&bNxt[bDst[i]]) = tb[i];
#endif
    }

    // 4 WMMA K-steps on the current chunk (A resident, B double-buffered)
    const int k0c = (c & 7) * KCHUNK;
#pragma unroll
    for (int ks = 0; ks < KCHUNK / 32; ++ks) {
      const int kb = ks * 32;
      v8h a0 = *reinterpret_cast<const v8h*>(pa + k0c + kb);
      v8h a1 = *reinterpret_cast<const v8h*>(pa + k0c + kb + 16);
      v8h b0 = *reinterpret_cast<const v8h*>(pb + kb);
      v8h b1 = *reinterpret_cast<const v8h*>(pb + kb + 16);
      v16h av = __builtin_shufflevector(a0, a1, 0, 1, 2, 3, 4, 5, 6, 7,
                                        8, 9, 10, 11, 12, 13, 14, 15);
      v16h bv = __builtin_shufflevector(b0, b1, 0, 1, 2, 3, 4, 5, 6, 7,
                                        8, 9, 10, 11, 12, 13, 14, 15);
      acc = __builtin_amdgcn_wmma_f32_16x16x32_f16(
          false, av, false, bv, (short)0, acc, false, false);
    }

    if ((c & 7) == 7) {
      // j-tile complete: distances + kNN update, all in registers
      const int j0 = (c >> 3) * NTILE;
      const int gj = j0 + nl;
#pragma unroll
      for (int r = 0; r < 8; ++r) {
        const float d2 = sqr[r] + sc - 2.0f * acc[r];
        float dd = __builtin_amdgcn_sqrtf(fmaxf(d2, 0.0f));
        if (gRowBase + r == gj) dd = BIGF;   // exclude self
        ins5(dd, l0[r], l1[r], l2[r], l3[r], l4[r]);
      }
      acc = (v8f){};
      const int jn = j0 + NTILE;
      if (jn < NPTS) sc = sq[jn + nl];       // pipelined 8 chunks ahead
    }

    wait_async_zero();   // loads for chunk c+1 ran under the WMMAs above
    __syncthreads();
  }

  // ---- merge 64 partial lists per row (LDS aliases the dead buffers) ----
  float* const parts = (float*)lds_raw;      // [MBLK][64*5 + 1] stride 321
#pragma unroll
  for (int r = 0; r < 8; ++r) {
    float* p = parts + (waveM * 16 + hiSel * 8 + r) * 321 + nl * 5;
    p[0] = l0[r]; p[1] = l1[r]; p[2] = l2[r]; p[3] = l3[r]; p[4] = l4[r];
  }
  __syncthreads();
  if (tid < MBLK) {
    float m0 = BIGF, m1 = BIGF, m2 = BIGF, m3 = BIGF, m4 = BIGF;
    const float* p = parts + tid * 321;
#pragma unroll 8
    for (int s = 0; s < 320; ++s) ins5(p[s], m0, m1, m2, m3, m4);
    const float score = (m0 + m1 + m2 + m3 + m4) * 0.2f;
    const float wv = (score > 0.1f) ? expf(-score * (1.0f / 32.0f)) : 0.0f;
    wout[rowBase + tid] = wv;
  }
}

// ---------------------------------------------------------------------------
// Kernel 3: out[0] = xr * w[row], out[1] = xi * w[row]
// ---------------------------------------------------------------------------
__global__ __launch_bounds__(256) void topo_scale_kernel(
    const float* __restrict__ xr, const float* __restrict__ xi,
    const float* __restrict__ w, float* __restrict__ out) {
  const int row = blockIdx.x;
  const int tid = threadIdx.x;
  const float wv = w[row];
  const size_t idx = (size_t)row * 256 + tid;
  const float4 a = reinterpret_cast<const float4*>(xr)[idx];
  const float4 b = reinterpret_cast<const float4*>(xi)[idx];
  float4 fa, fb;
  fa.x = a.x * wv; fa.y = a.y * wv; fa.z = a.z * wv; fa.w = a.w * wv;
  fb.x = b.x * wv; fb.y = b.y * wv; fb.z = b.z * wv; fb.w = b.w * wv;
  float4* o = reinterpret_cast<float4*>(out);
  o[idx] = fa;
  o[(size_t)NPTS * 256 + idx] = fb;
}

// ---------------------------------------------------------------------------
extern "C" void kernel_launch(void* const* d_in, const int* in_sizes, int n_in,
                              void* d_out, int out_size, void* d_ws, size_t ws_size,
                              hipStream_t stream) {
  (void)in_sizes; (void)n_in; (void)out_size; (void)ws_size;
  const float* xr = (const float*)d_in[0];
  const float* xi = (const float*)d_in[1];
  float* out = (float*)d_out;

  char* ws = (char*)d_ws;
  _Float16* xh = (_Float16*)ws;                        // 16 MB
  float* sq = (float*)(ws + (size_t)NPTS * DDIM * 2);  // 32 KB
  float* w  = sq + NPTS;                               // 32 KB

  topo_prep_kernel<<<NPTS, 256, 0, stream>>>(xr, xh, sq);
  topo_knn_kernel<<<NPTS / MBLK, 256, 0, stream>>>(xh, sq, w);
  topo_scale_kernel<<<NPTS, 256, 0, stream>>>(xr, xi, w, out);
}
